// GraphMAEEncoder_42236708389384
// MI455X (gfx1250) — matrix operand order repaired and checked by reading, not compiled, not run on hardware
//
#include <hip/hip_runtime.h>
#include <hip/hip_bf16.h>

#define N_NODES 50000
#define N_EDGES 800000
#define IN_D    64
#define HID     128
#define HEADS   4
#define CH      32
#define ET      (N_EDGES + N_NODES)   // self-loops appended
#define M_TILES (N_NODES / 16)        // 3125

typedef __attribute__((ext_vector_type(16))) __bf16         v16bf;
typedef __attribute__((ext_vector_type(8)))  float          v8f;
typedef __attribute__((ext_vector_type(8)))  unsigned short v8us;
typedef __attribute__((ext_vector_type(16))) unsigned short v16us;

union BFA { v16bf bf; struct { v8us lo, hi; } p; };
union BFB { v16bf bf; v16us u; };

__device__ __forceinline__ unsigned short f2bf(float f) {
  unsigned u = __float_as_uint(f);
  u += 0x7FFFu + ((u >> 16) & 1u);          // round-to-nearest-even
  return (unsigned short)(u >> 16);
}
__device__ __forceinline__ unsigned ford(float f) {   // order-preserving float->uint
  unsigned b = __float_as_uint(f);
  return b ^ ((b & 0x80000000u) ? 0xFFFFFFFFu : 0x80000000u);
}
__device__ __forceinline__ float fdec(unsigned o) {
  unsigned b = (o & 0x80000000u) ? (o ^ 0x80000000u) : ~o;
  return __uint_as_float(b);
}

// ---- weight transpose + bf16 pack: Wt[n][k] = bf16(W[k][n]) ----
__global__ void gmae_pack_wt(const float* __restrict__ W, unsigned short* __restrict__ Wt,
                             int K, int Nout) {
  int i = blockIdx.x * blockDim.x + threadIdx.x;
  if (i >= K * Nout) return;
  int k = i % K, n = i / K;
  Wt[(size_t)n * K + k] = f2bf(W[(size_t)k * Nout + n]);
}

__global__ void gmae_to_bf16(const float* __restrict__ s, unsigned short* __restrict__ d, int n) {
  int i = blockIdx.x * blockDim.x + threadIdx.x;
  if (i < n) d[i] = f2bf(s[i]);
}

// ---- WMMA GEMM: C[M,Nout] = A[M,K](bf16) * Bt[Nout,K]^T (+bias, + optional bf16 copy) ----
// block = 256 (8 waves). Block computes 128 rows x 64 cols.
// B tile (64 cols x K) staged in LDS via gfx1250 async copy; each wave: 1 A-frag -> 4 WMMAs.
__global__ void gmae_gemm_wmma(const unsigned short* __restrict__ A,
                               const unsigned short* __restrict__ Bt,
                               const float* __restrict__ bias,
                               float* __restrict__ Co,
                               unsigned short* __restrict__ Co_bf,
                               int K, int Nout) {
  __shared__ __align__(32) unsigned short sB[64 * (HID + 8)];  // padded stride kills bank conflicts
  const int SK  = K + 8;
  const int tid = threadIdx.x;
  const int n0  = blockIdx.y * 64;

  // ---- stage Bt[n0 .. n0+63][0..K) into LDS with gfx1250 async-LDS loads ----
  {
    const int rowchunks = K / 8;                 // 16B chunks per row
    const int chunks    = 64 * rowchunks;        // 512 (K=64) or 1024 (K=128)
    for (int c = tid; c < chunks; c += 256) {
      int n = c / rowchunks, j = c - n * rowchunks;
      unsigned           loff  = (unsigned)(unsigned long long)(&sB[n * SK + j * 8]);
      unsigned long long gaddr = (unsigned long long)(Bt + (size_t)(n0 + n) * K + j * 8);
      asm volatile("global_load_async_to_lds_b128 %0, %1, off"
                   :: "v"(loff), "v"(gaddr) : "memory");
    }
    asm volatile("s_wait_asynccnt 0" ::: "memory");
    __syncthreads();
  }

  const int wave   = tid >> 5;
  const int lane   = tid & 31;
  const int hi     = lane >> 4;
  const int nl     = lane & 15;
  const int m_tile = blockIdx.x * 8 + wave;
  if (m_tile >= M_TILES) return;                  // wave-uniform: EXEC stays all-1s
  const int m_base = m_tile * 16;
  const int row    = m_base + nl;

  v8f acc0 = {}, acc1 = {}, acc2 = {}, acc3 = {};
  for (int k0 = 0; k0 < K; k0 += 32) {
    BFA a;
    const int kA = k0 + (hi ? 16 : 0);            // contiguous k layout (matches B)
    a.p.lo = *(const v8us*)(A + (size_t)row * K + kA);
    a.p.hi = *(const v8us*)(A + (size_t)row * K + kA + 8);
    const int kB = k0 + (hi ? 16 : 0);            // contiguous k layout (matches A)
    BFB b0, b1, b2, b3;
    b0.u = *(const v16us*)(&sB[(0 * 16 + nl) * SK + kB]);
    b1.u = *(const v16us*)(&sB[(1 * 16 + nl) * SK + kB]);
    b2.u = *(const v16us*)(&sB[(2 * 16 + nl) * SK + kB]);
    b3.u = *(const v16us*)(&sB[(3 * 16 + nl) * SK + kB]);
    acc0 = __builtin_amdgcn_wmma_f32_16x16x32_bf16(false, a.bf, false, b0.bf, (short)0, acc0, false, false);
    acc1 = __builtin_amdgcn_wmma_f32_16x16x32_bf16(false, a.bf, false, b1.bf, (short)0, acc1, false, false);
    acc2 = __builtin_amdgcn_wmma_f32_16x16x32_bf16(false, a.bf, false, b2.bf, (short)0, acc2, false, false);
    acc3 = __builtin_amdgcn_wmma_f32_16x16x32_bf16(false, a.bf, false, b3.bf, (short)0, acc3, false, false);
  }

  const int rbase = m_base + (hi ? 8 : 0);
  v8f accs[4] = {acc0, acc1, acc2, acc3};
#pragma unroll
  for (int t = 0; t < 4; ++t) {
    const int col = n0 + t * 16 + nl;
    const float bv = bias ? bias[col] : 0.f;
#pragma unroll
    for (int r = 0; r < 8; ++r) {
      float y = accs[t][r] + bv;
      Co[(size_t)(rbase + r) * Nout + col] = y;
      if (Co_bf) Co_bf[(size_t)(rbase + r) * Nout + col] = f2bf(y);
    }
  }
}

// ---- per-(node,head) attention logits ----
__global__ void gmae_scores(const float* __restrict__ xh, const float* __restrict__ asrc,
                            const float* __restrict__ adst, float* __restrict__ ssrc,
                            float* __restrict__ sdst) {
  int i = blockIdx.x * blockDim.x + threadIdx.x;   // (node*HEADS + head)
  if (i >= N_NODES * HEADS) return;
  int h = i & (HEADS - 1);
  const float* v  = xh + (size_t)(i >> 2) * HID + h * CH;
  const float* as = asrc + h * CH;
  const float* ad = adst + h * CH;
  float s0 = 0.f, s1 = 0.f;
#pragma unroll
  for (int c = 0; c < CH; c += 4) {
    float4 xv = *(const float4*)(v + c);
    float4 av = *(const float4*)(as + c);
    float4 dv = *(const float4*)(ad + c);
    s0 += xv.x * av.x + xv.y * av.y + xv.z * av.z + xv.w * av.w;
    s1 += xv.x * dv.x + xv.y * dv.y + xv.z * dv.z + xv.w * dv.w;
  }
  ssrc[i] = s0; sdst[i] = s1;
}

__device__ __forceinline__ void edge_sd(const int* ei, int e, int& s, int& d) {
  if (e < N_EDGES) { s = ei[e]; d = ei[N_EDGES + e]; }
  else             { s = e - N_EDGES; d = s; }
}
__device__ __forceinline__ float lrelu(float v) { return v > 0.f ? v : 0.2f * v; }

__global__ void gmae_edge_max(const int* __restrict__ ei, const float* __restrict__ ssrc,
                              const float* __restrict__ sdst, unsigned* __restrict__ mo) {
  int e = blockIdx.x * blockDim.x + threadIdx.x;
  if (e >= ET) return;
  int s, d; edge_sd(ei, e, s, d);
  float4 a = *(const float4*)(ssrc + (size_t)s * HEADS);
  float4 b = *(const float4*)(sdst + (size_t)d * HEADS);
  atomicMax(&mo[d * HEADS + 0], ford(lrelu(a.x + b.x)));
  atomicMax(&mo[d * HEADS + 1], ford(lrelu(a.y + b.y)));
  atomicMax(&mo[d * HEADS + 2], ford(lrelu(a.z + b.z)));
  atomicMax(&mo[d * HEADS + 3], ford(lrelu(a.w + b.w)));
}

__global__ void gmae_edge_sum(const int* __restrict__ ei, const float* __restrict__ ssrc,
                              const float* __restrict__ sdst, const unsigned* __restrict__ mo,
                              float* __restrict__ z) {
  int e = blockIdx.x * blockDim.x + threadIdx.x;
  if (e >= ET) return;
  int s, d; edge_sd(ei, e, s, d);
  float4 a = *(const float4*)(ssrc + (size_t)s * HEADS);
  float4 b = *(const float4*)(sdst + (size_t)d * HEADS);
  uint4  m = *(const uint4*)(mo + (size_t)d * HEADS);
  atomicAdd(&z[d * HEADS + 0], __expf(lrelu(a.x + b.x) - fdec(m.x)));
  atomicAdd(&z[d * HEADS + 1], __expf(lrelu(a.y + b.y) - fdec(m.y)));
  atomicAdd(&z[d * HEADS + 2], __expf(lrelu(a.z + b.z) - fdec(m.z)));
  atomicAdd(&z[d * HEADS + 3], __expf(lrelu(a.w + b.w) - fdec(m.w)));
}

// one wave per edge; lanes 0-3 compute per-head alpha, broadcast via shfl;
// each lane scatters 4 channels with f32 atomics (L2-resident accumulator)
__global__ void gmae_edge_msg(const int* __restrict__ ei, const float* __restrict__ ssrc,
                              const float* __restrict__ sdst, const unsigned* __restrict__ mo,
                              const float* __restrict__ z, const float* __restrict__ xh,
                              float* __restrict__ out) {
  int t = blockIdx.x * blockDim.x + threadIdx.x;
  int e = t >> 5;
  if (e >= ET) return;
  int lane = t & 31;
  int s, d; edge_sd(ei, e, s, d);
  float al = 0.f;
  if (lane < HEADS) {
    float v = lrelu(ssrc[s * HEADS + lane] + sdst[d * HEADS + lane]);
    float p = __expf(v - fdec(mo[d * HEADS + lane]));
    al = p / (z[d * HEADS + lane] + 1e-16f);
  }
  int   h     = lane >> 3;                     // 8 lanes per head
  float alpha = __shfl(al, h, 32);
  int   c     = lane * 4;
  const float4 xv = *(const float4*)(xh + (size_t)s * HID + c);
  float* o = out + (size_t)d * HID + c;
  atomicAdd(o + 0, xv.x * alpha);
  atomicAdd(o + 1, xv.y * alpha);
  atomicAdd(o + 2, xv.z * alpha);
  atomicAdd(o + 3, xv.w * alpha);
}

// ---- fused residual + bias + LayerNorm; wave per node ----
__global__ void gmae_ln(const float* __restrict__ h, const float* __restrict__ gout,
                        const float* __restrict__ gb, const float* __restrict__ g,
                        const float* __restrict__ b, float* __restrict__ out,
                        unsigned short* __restrict__ out_bf) {
  int n = blockIdx.x * blockDim.y + threadIdx.y;
  if (n >= N_NODES) return;
  int lane = threadIdx.x, c = lane * 4;
  size_t base = (size_t)n * HID + c;
  float4 hv = *(const float4*)(h + base);
  float4 gv = *(const float4*)(gout + base);
  float4 bb = *(const float4*)(gb + c);
  float x0 = hv.x + gv.x + bb.x, x1 = hv.y + gv.y + bb.y;
  float x2 = hv.z + gv.z + bb.z, x3 = hv.w + gv.w + bb.w;
  float sum = x0 + x1 + x2 + x3;
  float sq  = x0 * x0 + x1 * x1 + x2 * x2 + x3 * x3;
#pragma unroll
  for (int m = 16; m >= 1; m >>= 1) {
    sum += __shfl_xor(sum, m, 32);
    sq  += __shfl_xor(sq,  m, 32);
  }
  float mu  = sum * (1.f / HID);
  float var = sq * (1.f / HID) - mu * mu;
  float inv = rsqrtf(var + 1e-5f);
  float4 gg = *(const float4*)(g + c);
  float4 b2 = *(const float4*)(b + c);
  float y0 = (x0 - mu) * inv * gg.x + b2.x;
  float y1 = (x1 - mu) * inv * gg.y + b2.y;
  float y2 = (x2 - mu) * inv * gg.z + b2.z;
  float y3 = (x3 - mu) * inv * gg.w + b2.w;
  *(float4*)(out + base) = make_float4(y0, y1, y2, y3);
  if (out_bf) {
    out_bf[base + 0] = f2bf(y0); out_bf[base + 1] = f2bf(y1);
    out_bf[base + 2] = f2bf(y2); out_bf[base + 3] = f2bf(y3);
  }
}

extern "C" void kernel_launch(void* const* d_in, const int* in_sizes, int n_in,
                              void* d_out, int out_size, void* d_ws, size_t ws_size,
                              hipStream_t stream) {
  const float* x     = (const float*)d_in[0];
  const int*   ei    = (const int*)d_in[1];
  const float* W_in  = (const float*)d_in[2];
  const float* b_in  = (const float*)d_in[3];
  const float* gW[2]  = {(const float*)d_in[4],  (const float*)d_in[10]};
  const float* gas[2] = {(const float*)d_in[5],  (const float*)d_in[11]};
  const float* gad[2] = {(const float*)d_in[6],  (const float*)d_in[12]};
  const float* gb[2]  = {(const float*)d_in[7],  (const float*)d_in[13]};
  const float* lg[2]  = {(const float*)d_in[8],  (const float*)d_in[14]};
  const float* lb[2]  = {(const float*)d_in[9],  (const float*)d_in[15]};

  char* w = (char*)d_ws;
  float*          A    = (float*)w;          w += (size_t)N_NODES * HID * 4;  // node state f32
  unsigned short* Bh   = (unsigned short*)w; w += (size_t)N_NODES * HID * 2;  // bf16 state
  float*          Cx   = (float*)w;          w += (size_t)N_NODES * HID * 4;  // xh
  float*          Dacc = (float*)w;          w += (size_t)N_NODES * HID * 4;  // scatter accum
  float*          ssrc = (float*)w;          w += (size_t)N_NODES * HEADS * 4;
  float*          sdst = (float*)w;          w += (size_t)N_NODES * HEADS * 4;
  unsigned*       mo   = (unsigned*)w;       w += (size_t)N_NODES * HEADS * 4;
  float*          z    = (float*)w;          w += (size_t)N_NODES * HEADS * 4;
  unsigned short* WtIn = (unsigned short*)w; w += (size_t)HID * IN_D * 2;
  unsigned short* Wt[2];
  Wt[0] = (unsigned short*)w;                w += (size_t)HID * HID * 2;
  Wt[1] = (unsigned short*)w;                w += (size_t)HID * HID * 2;

  // pack weights (transposed, bf16)
  gmae_pack_wt<<<(IN_D * HID + 255) / 256, 256, 0, stream>>>(W_in, WtIn, IN_D, HID);
  gmae_pack_wt<<<(HID * HID + 255) / 256, 256, 0, stream>>>(gW[0], Wt[0], HID, HID);
  gmae_pack_wt<<<(HID * HID + 255) / 256, 256, 0, stream>>>(gW[1], Wt[1], HID, HID);

  const dim3 ggrid((M_TILES + 7) / 8, HID / 64);

  // input projection: h0 = x @ W_in + b_in.
  // NOTE: Bh is this GEMM's bf16 *input* (K=64 layout); fused bf16 epilogue must stay
  // disabled here (would race with readers of the same buffer). The bf16 state for the
  // next GEMM is regenerated by the separate pass below.
  gmae_to_bf16<<<(N_NODES * IN_D + 255) / 256, 256, 0, stream>>>(x, Bh, N_NODES * IN_D);
  gmae_gemm_wmma<<<ggrid, 256, 0, stream>>>(Bh, WtIn, b_in, A, nullptr, IN_D, HID);
  gmae_to_bf16<<<(N_NODES * HID + 255) / 256, 256, 0, stream>>>(A, Bh, N_NODES * HID);

  const int egrid = (ET + 255) / 256;
  const int mgrid = (ET * 32 + 255) / 256;
  const int sgrid = (N_NODES * HEADS + 255) / 256;

  for (int L = 0; L < 2; ++L) {
    gmae_gemm_wmma<<<ggrid, 256, 0, stream>>>(Bh, Wt[L], nullptr, Cx, nullptr, HID, HID);
    gmae_scores<<<sgrid, 256, 0, stream>>>(Cx, gas[L], gad[L], ssrc, sdst);
    hipMemsetAsync(mo, 0, (size_t)N_NODES * HEADS * 4, stream);
    hipMemsetAsync(z, 0, (size_t)N_NODES * HEADS * 4, stream);
    hipMemsetAsync(Dacc, 0, (size_t)N_NODES * HID * 4, stream);
    gmae_edge_max<<<egrid, 256, 0, stream>>>(ei, ssrc, sdst, mo);
    gmae_edge_sum<<<egrid, 256, 0, stream>>>(ei, ssrc, sdst, mo, z);
    gmae_edge_msg<<<mgrid, 256, 0, stream>>>(ei, ssrc, sdst, mo, z, Cx, Dacc);
    float* outp = (L == 1) ? (float*)d_out : A;
    unsigned short* bfp = (L == 1) ? nullptr : Bh;
    gmae_ln<<<(N_NODES + 3) / 4, dim3(32, 4), 0, stream>>>(
        A, Dacc, gb[L], lg[L], lb[L], outp, bfp);
  }
}
